// SelfAttention_21964462751900
// MI455X (gfx1250) — matrix operand order repaired and checked by reading, not compiled, not run on hardware
//
#include <hip/hip_runtime.h>
#include <hip/hip_bf16.h>

// ---------------------------------------------------------------------------
// Fused multi-head self-attention for MI455X (gfx1250, wave32, WMMA + TDM).
// - all matmuls: v_wmma_f32_16x16x32_f16 (f16 in, fp32 accumulate)
// - flash-attention: (B,H,L,L) scores never materialized
// - K/V tiles staged into LDS by the Tensor Data Mover (tensor_load_to_lds),
//   double-buffered, completion via s_wait_tensorcnt + workgroup barrier
// ---------------------------------------------------------------------------

#define B_  2
#define L_  2048
#define H_  16
#define DH_ 64
#define DM_ 1024
#define KSTEP 32
#define INV_SCALER 0.25f   // 1/sqrt(N_HEADS) = 1/4

typedef _Float16 v8h  __attribute__((ext_vector_type(8)));
typedef _Float16 v16h __attribute__((ext_vector_type(16)));
typedef float    v8f  __attribute__((ext_vector_type(8)));
typedef unsigned int u32x4 __attribute__((ext_vector_type(4)));
typedef int          i32x8 __attribute__((ext_vector_type(8)));
typedef int          i32x4 __attribute__((ext_vector_type(4)));

__device__ __forceinline__ v8f wmma16(v16h a, v16h b, v8f c) {
  return __builtin_amdgcn_wmma_f32_16x16x32_f16(false, a, false, b, (short)0, c,
                                                false, false);
}

// A-operand (16x32 f16): lane (m,half): e=0..7 <-> K=8h.., e=8..15 <-> K=16+8h..
__device__ __forceinline__ v16h load_a_f16(const _Float16* base, int half) {
  const v8h c0 = *(const v8h*)(base + 8 * half);
  const v8h c1 = *(const v8h*)(base + 16 + 8 * half);
  v16h r;
#pragma unroll
  for (int e = 0; e < 8; ++e) { r[e] = c0[e]; r[8 + e] = c1[e]; }
  return r;
}

__device__ __forceinline__ v16h load_a_f32(const float* base, int half) {
  const float4 x0 = *(const float4*)(base + 8 * half);
  const float4 x1 = *(const float4*)(base + 8 * half + 4);
  const float4 y0 = *(const float4*)(base + 16 + 8 * half);
  const float4 y1 = *(const float4*)(base + 16 + 8 * half + 4);
  v16h r;
  r[0]  = (_Float16)x0.x; r[1]  = (_Float16)x0.y; r[2]  = (_Float16)x0.z; r[3]  = (_Float16)x0.w;
  r[4]  = (_Float16)x1.x; r[5]  = (_Float16)x1.y; r[6]  = (_Float16)x1.z; r[7]  = (_Float16)x1.w;
  r[8]  = (_Float16)y0.x; r[9]  = (_Float16)y0.y; r[10] = (_Float16)y0.z; r[11] = (_Float16)y0.w;
  r[12] = (_Float16)y1.x; r[13] = (_Float16)y1.y; r[14] = (_Float16)y1.z; r[15] = (_Float16)y1.w;
  return r;
}

// B-operand (32x16 f16): lane (n,half): e=0..7 <-> K=8h.., e=8..15 <-> K=16+8h..
__device__ __forceinline__ v16h load_b_f16(const _Float16* base, int half) {
  const v8h c0 = *(const v8h*)(base + 8 * half);
  const v8h c1 = *(const v8h*)(base + 16 + 8 * half);
  v16h r;
#pragma unroll
  for (int e = 0; e < 8; ++e) { r[e] = c0[e]; r[8 + e] = c1[e]; }
  return r;
}

// Generic pointer to LDS: low 32 bits of the flat address are the LDS offset.
__device__ __forceinline__ unsigned lds_off(const void* p) {
  return (unsigned)(size_t)p;
}

// ---------------------------------------------------------------------------
// Tensor Data Mover: 2-D tile load (f16 elements, data_size code 1 = 2 bytes).
// Descriptor per CDNA5 ISA (groups 0/1; groups 2/3 and
// the trailing group are zero: tile_dim2 = 0 -> pure 2-D tile). Uniform args;
// issued once per wave; completion tracked with TENSORcnt.
// ---------------------------------------------------------------------------
__device__ __forceinline__ void tdm_load_2d_f16(
    unsigned lds_byte_off, const _Float16* gaddr,
    unsigned tensor_d0, unsigned tensor_d1,
    unsigned tile_d0, unsigned tile_d1,
    unsigned stride_d0 /* in elements */) {
  const unsigned long long ga = (unsigned long long)(size_t)gaddr;
  u32x4 g0;
  g0[0] = 1u;                                   // count=1, user descriptor
  g0[1] = lds_byte_off;                         // lds_addr [63:32]
  g0[2] = (unsigned)(ga & 0xffffffffu);         // global_addr[31:0]
  g0[3] = (unsigned)((ga >> 32) & 0x1ffffffu)   // global_addr[56:32]
        | (2u << 30);                           // type = 2 ("image")
  i32x8 g1;
  g1[0] = (int)(1u << 16);                      // wg_mask=0, data_size=1 (2B)
  g1[1] = (int)((tensor_d0 & 0xffffu) << 16);   // tensor_dim0[15:0] @ 63:48
  g1[2] = (int)(((tensor_d0 >> 16) & 0xffffu) | ((tensor_d1 & 0xffffu) << 16));
  g1[3] = (int)(((tensor_d1 >> 16) & 0xffffu) | ((tile_d0 & 0xffffu) << 16));
  g1[4] = (int)(tile_d1 & 0xffffu);             // tile_dim1; tile_dim2=0 (2-D)
  g1[5] = (int)stride_d0;                       // tensor_dim0_stride[31:0]
  g1[6] = 0;
  g1[7] = 0;
  const i32x4 z4 = {0, 0, 0, 0};                // groups 2/3 unused (2-D)
  const i32x8 z8 = {0, 0, 0, 0, 0, 0, 0, 0};    // trailing group (clang-23 form)
  __builtin_amdgcn_tensor_load_to_lds(g0, g1, z4, z4, z8, 0);
}

// ---------------------------------------------------------------------------
// Kernel 1: fp32 -> f16 weight conversion.
// ---------------------------------------------------------------------------
__global__ void cvt_weights_kernel(const float* __restrict__ Wq,
                                   const float* __restrict__ Wk,
                                   const float* __restrict__ Wv,
                                   const float* __restrict__ Wo,
                                   _Float16* __restrict__ whq,
                                   _Float16* __restrict__ whk,
                                   _Float16* __restrict__ whv,
                                   _Float16* __restrict__ who) {
  const int i = blockIdx.x * blockDim.x + threadIdx.x;
  if (i < DH_ * DH_) {
    whq[i] = (_Float16)Wq[i];
    whk[i] = (_Float16)Wk[i];
    whv[i] = (_Float16)Wv[i];
  }
  if (i < DM_ * DM_) who[i] = (_Float16)Wo[i];
}

// ---------------------------------------------------------------------------
// Kernel 2: per-head projection y = x_h @ W^T. One wave -> 16(L) x 64(e) tile.
// ---------------------------------------------------------------------------
__global__ __launch_bounds__(32) void qkv_proj_kernel(
    const float* __restrict__ X, const _Float16* __restrict__ W,
    _Float16* __restrict__ outRM, _Float16* __restrict__ outTR, int transposed) {
  const int lane = threadIdx.x;
  const int half = lane >> 4;
  const int ln   = lane & 15;
  const int l0   = blockIdx.x * 16;
  const int h    = blockIdx.y;
  const int b    = blockIdx.z;

  const float* xrow = X + ((size_t)b * L_ + (l0 + ln)) * DM_ + h * DH_;
  const v16h a0 = load_a_f32(xrow, half);
  const v16h a1 = load_a_f32(xrow + 32, half);

  v8f acc[4];
#pragma unroll
  for (int j = 0; j < 4; ++j)
#pragma unroll
    for (int r = 0; r < 8; ++r) acc[j][r] = 0.f;

#pragma unroll
  for (int j = 0; j < 4; ++j) {
    const _Float16* wrow = W + (size_t)(16 * j + ln) * DH_;
    acc[j] = wmma16(a0, load_b_f16(wrow, half), acc[j]);
    acc[j] = wmma16(a1, load_b_f16(wrow + 32, half), acc[j]);
  }

  const size_t bh = (size_t)b * H_ + h;
  if (!transposed) {
#pragma unroll
    for (int j = 0; j < 4; ++j)
#pragma unroll
      for (int r = 0; r < 8; ++r)
        outRM[(bh * L_ + (l0 + r + 8 * half)) * DH_ + 16 * j + ln] =
            (_Float16)acc[j][r];
  } else {
#pragma unroll
    for (int j = 0; j < 4; ++j) {
      v8h pk;
#pragma unroll
      for (int r = 0; r < 8; ++r) pk[r] = (_Float16)acc[j][r];
      *(v8h*)(outTR + (bh * DH_ + 16 * j + ln) * L_ + l0 + 8 * half) = pk;
    }
  }
}

// ---------------------------------------------------------------------------
// Kernel 3: fused flash attention.
// Block = 4 waves = 64 queries; wave w owns q-tile q0 = 64*blk + 16*w.
// K (32x64) and V^T (64x32) tiles are TDM-loaded into LDS, double-buffered;
// all four waves consume them (4x L2-traffic reduction vs 1-wave blocks).
// ---------------------------------------------------------------------------
__global__ __launch_bounds__(128) void attn_kernel(
    const _Float16* __restrict__ qp, const _Float16* __restrict__ kp,
    const _Float16* __restrict__ vt, const int* __restrict__ mask,
    _Float16* __restrict__ oc) {
  __shared__ _Float16 lds_k[2][KSTEP * DH_];   // keys x 64, row-major
  __shared__ _Float16 lds_v[2][DH_ * KSTEP];   // d x keys, row-major
  __shared__ _Float16 lds_p[4][16 * 32];       // per-wave P exchange tile

  const int tid  = threadIdx.x;
  const int wave = tid >> 5;
  const int lane = tid & 31;
  const int half = lane >> 4;
  const int ln   = lane & 15;
  const int q0   = blockIdx.x * 64 + wave * 16;
  const int h    = blockIdx.y;
  const int b    = blockIdx.z;
  const size_t bh = (size_t)b * H_ + h;

  const _Float16* kbase = kp + bh * (size_t)L_ * DH_;  // (L, 64)
  const _Float16* vbase = vt + bh * (size_t)DH_ * L_;  // (64, L)

  const _Float16* qrow = qp + (bh * L_ + q0 + ln) * DH_;
  const v16h qa0 = load_a_f16(qrow, half);
  const v16h qa1 = load_a_f16(qrow + 32, half);

  v8f acc[4];
#pragma unroll
  for (int j = 0; j < 4; ++j)
#pragma unroll
    for (int r = 0; r < 8; ++r) acc[j][r] = 0.f;

  float mrun[8], lrun[8];
#pragma unroll
  for (int r = 0; r < 8; ++r) { mrun[r] = -1e30f; lrun[r] = 0.f; }

  const int* maskbase = mask + ((size_t)b * L_ + q0) * L_;  // (B,1,L,L)

  // Prologue: TDM-load first K/V tiles (wave 0 issues; DMA runs async).
  if (wave == 0) {
    tdm_load_2d_f16(lds_off(&lds_k[0][0]), kbase, DH_, L_, DH_, KSTEP, DH_);
    tdm_load_2d_f16(lds_off(&lds_v[0][0]), vbase, L_, DH_, KSTEP, DH_, L_);
  }

  for (int kk = 0; kk < L_; kk += KSTEP) {
    const int cur = (kk / KSTEP) & 1;

    if (wave == 0) __builtin_amdgcn_s_wait_tensorcnt(0);
    __syncthreads();  // publish TDM data; also fences last step's ds reads

    // Prefetch next K/V tiles into the other buffer while we compute.
    if (wave == 0 && (kk + KSTEP) < L_) {
      tdm_load_2d_f16(lds_off(&lds_k[cur ^ 1][0]),
                      kbase + (size_t)(kk + KSTEP) * DH_,
                      DH_, L_, DH_, KSTEP, DH_);
      tdm_load_2d_f16(lds_off(&lds_v[cur ^ 1][0]),
                      vbase + (kk + KSTEP),
                      L_, DH_, KSTEP, DH_, L_);
    }

    const _Float16* lk = &lds_k[cur][0];
    const _Float16* lv = &lds_v[cur][0];

    // ---- scores: S = Q K^T for keys kk..kk+31 (two 16x16 tiles) ----
    v8f s0, s1;
#pragma unroll
    for (int r = 0; r < 8; ++r) { s0[r] = 0.f; s1[r] = 0.f; }
    {
      const _Float16* krow0 = lk + (size_t)ln * DH_;
      s0 = wmma16(qa0, load_b_f16(krow0, half), s0);
      s0 = wmma16(qa1, load_b_f16(krow0 + 32, half), s0);
      const _Float16* krow1 = lk + (size_t)(16 + ln) * DH_;
      s1 = wmma16(qa0, load_b_f16(krow1, half), s1);
      s1 = wmma16(qa1, load_b_f16(krow1 + 32, half), s1);
    }

    // ---- scale, mask, online softmax (rows live in 16-lane halves) ----
    float alpha[8];
#pragma unroll
    for (int r = 0; r < 8; ++r) {
      const int row = r + 8 * half;
      float x0 = s0[r] * INV_SCALER;
      float x1 = s1[r] * INV_SCALER;
      if (maskbase[(size_t)row * L_ + kk + ln] == 1)      x0 = -1e30f;
      if (maskbase[(size_t)row * L_ + kk + 16 + ln] == 1) x1 = -1e30f;

      float mx = fmaxf(x0, x1);
      mx = fmaxf(mx, __shfl_xor(mx, 1));
      mx = fmaxf(mx, __shfl_xor(mx, 2));
      mx = fmaxf(mx, __shfl_xor(mx, 4));
      mx = fmaxf(mx, __shfl_xor(mx, 8));
      mx = fmaxf(mx, mrun[r]);

      const float al = __expf(mrun[r] - mx);
      const float p0 = __expf(x0 - mx);
      const float p1 = __expf(x1 - mx);
      float ps = p0 + p1;
      ps += __shfl_xor(ps, 1);
      ps += __shfl_xor(ps, 2);
      ps += __shfl_xor(ps, 4);
      ps += __shfl_xor(ps, 8);

      lrun[r]  = lrun[r] * al + ps;
      mrun[r]  = mx;
      alpha[r] = al;

      lds_p[wave][row * 32 + ln]      = (_Float16)p0;
      lds_p[wave][row * 32 + 16 + ln] = (_Float16)p1;
    }

#pragma unroll
    for (int j = 0; j < 4; ++j)
#pragma unroll
      for (int r = 0; r < 8; ++r) acc[j][r] *= alpha[r];

    // P exchange is wave-private: LDS is in-order per wave; the wait+clobber
    // stops the compiler reordering the cross-lane ds store->load pattern.
    asm volatile("s_wait_dscnt 0x0" ::: "memory");

    // ---- O += P V : P in A layout from LDS, V^T rows as B operand ----
    const v16h pa = load_a_f16(&lds_p[wave][ln * 32], half);
#pragma unroll
    for (int j = 0; j < 4; ++j) {
      const _Float16* vrow = lv + (size_t)(16 * j + ln) * KSTEP;
      acc[j] = wmma16(pa, load_b_f16(vrow, half), acc[j]);
    }
  }

  // ---- normalize and store concat head outputs: (B, L, H*64) f16 ----
#pragma unroll
  for (int j = 0; j < 4; ++j)
#pragma unroll
    for (int r = 0; r < 8; ++r) {
      const float o = acc[j][r] / lrun[r];
      oc[((size_t)b * L_ + q0 + r + 8 * half) * DM_ + h * DH_ + 16 * j + ln] =
          (_Float16)o;
    }
}

// ---------------------------------------------------------------------------
// Kernel 4: output projection Y = Oc @ Wo^T + bo.
// Block = 4 waves sharing one TDM-staged 16x1024 A-tile in LDS;
// wave w computes the 64-column panel c = 256*blockIdx.y + 64*w.
// ---------------------------------------------------------------------------
__global__ __launch_bounds__(128) void oproj_kernel(
    const _Float16* __restrict__ oc, const _Float16* __restrict__ who,
    const float* __restrict__ bo, float* __restrict__ out) {
  __shared__ _Float16 lds_a[16 * DM_];  // 32 KB

  const int tid  = threadIdx.x;
  const int wave = tid >> 5;
  const int lane = tid & 31;
  const int half = lane >> 4;
  const int ln   = lane & 15;
  const int r0   = blockIdx.x * 16;              // rows over B*L
  const int c0   = blockIdx.y * 256 + wave * 64; // cols over 1024

  if (wave == 0) {
    tdm_load_2d_f16(lds_off(&lds_a[0]), oc + (size_t)r0 * DM_,
                    DM_, B_ * L_, DM_, 16, DM_);
    __builtin_amdgcn_s_wait_tensorcnt(0);
  }
  __syncthreads();

  const _Float16* arow = &lds_a[(size_t)ln * DM_];

  v8f acc[4];
#pragma unroll
  for (int j = 0; j < 4; ++j)
#pragma unroll
    for (int r = 0; r < 8; ++r) acc[j][r] = 0.f;

  for (int k = 0; k < DM_; k += 32) {
    const v16h a = load_a_f16(arow + k, half);
#pragma unroll
    for (int j = 0; j < 4; ++j) {
      const _Float16* wrow = who + (size_t)(c0 + 16 * j + ln) * DM_ + k;
      acc[j] = wmma16(a, load_b_f16(wrow, half), acc[j]);
    }
  }

#pragma unroll
  for (int j = 0; j < 4; ++j) {
    const float bias = bo[c0 + 16 * j + ln];
#pragma unroll
    for (int r = 0; r < 8; ++r)
      out[((size_t)r0 + r + 8 * half) * DM_ + c0 + 16 * j + ln] =
          acc[j][r] + bias;
  }
}

// ---------------------------------------------------------------------------
// Host launcher. Workspace ~34 MB: f16 weights, projected q/k, v^T, concat out.
// ---------------------------------------------------------------------------
extern "C" void kernel_launch(void* const* d_in, const int* in_sizes, int n_in,
                              void* d_out, int out_size, void* d_ws, size_t ws_size,
                              hipStream_t stream) {
  (void)in_sizes; (void)n_in; (void)out_size; (void)ws_size;

  const float* query = (const float*)d_in[0];
  const float* key_t = (const float*)d_in[1];
  const float* value = (const float*)d_in[2];
  const int*   mask  = (const int*)d_in[3];
  const float* Wq    = (const float*)d_in[4];
  const float* Wk    = (const float*)d_in[5];
  const float* Wv    = (const float*)d_in[6];
  const float* Wo    = (const float*)d_in[7];
  const float* bo    = (const float*)d_in[8];
  float* out = (float*)d_out;

  char* ws = (char*)d_ws;
  size_t off = 0;
  auto carve = [&](size_t bytes) -> char* {
    char* p = ws + off;
    off += (bytes + 255) & ~(size_t)255;
    return p;
  };
  _Float16* whq = (_Float16*)carve((size_t)DH_ * DH_ * 2);
  _Float16* whk = (_Float16*)carve((size_t)DH_ * DH_ * 2);
  _Float16* whv = (_Float16*)carve((size_t)DH_ * DH_ * 2);
  _Float16* who = (_Float16*)carve((size_t)DM_ * DM_ * 2);
  _Float16* qp  = (_Float16*)carve((size_t)B_ * H_ * L_ * DH_ * 2);  // (B,H,L,64)
  _Float16* kp  = (_Float16*)carve((size_t)B_ * H_ * L_ * DH_ * 2);  // (B,H,L,64)
  _Float16* vt  = (_Float16*)carve((size_t)B_ * H_ * DH_ * L_ * 2);  // (B,H,64,L)
  _Float16* oc  = (_Float16*)carve((size_t)B_ * L_ * DM_ * 2);       // (B,L,1024)

  cvt_weights_kernel<<<dim3((DM_ * DM_) / 256), dim3(256), 0, stream>>>(
      Wq, Wk, Wv, Wo, whq, whk, whv, who);

  const dim3 pgrid(L_ / 16, H_, B_);
  qkv_proj_kernel<<<pgrid, 32, 0, stream>>>(query, whq, qp, nullptr, 0);
  qkv_proj_kernel<<<pgrid, 32, 0, stream>>>(key_t, whk, kp, nullptr, 0);
  qkv_proj_kernel<<<pgrid, 32, 0, stream>>>(value, whv, nullptr, vt, 1);

  attn_kernel<<<dim3(L_ / 64, H_, B_), 128, 0, stream>>>(qp, kp, vt, mask, oc);

  oproj_kernel<<<dim3((B_ * L_) / 16, DM_ / 256), 128, 0, stream>>>(oc, who, bo, out);
}
